// MultiHeadLinearSelfAttention_78804059947467
// MI455X (gfx1250) — matrix operand order repaired and checked by reading, not compiled, not run on hardware
//
#include <hip/hip_runtime.h>
#include <hip/hip_bf16.h>
#include <cstdint>

typedef __bf16 bf16;
typedef __attribute__((ext_vector_type(16))) __bf16 v16bf;
typedef __attribute__((ext_vector_type(8)))  float  v8f;

#define BATCH 16
#define CDIM  256
#define NHEAD 8
#define HD    32
#define NPIX  4096
#define IMGW  64
#define OC3   768

union Frag16 { v16bf v; uint4 q[2]; };
union Pack8  { bf16 h[8]; uint4 u; };

static __device__ __forceinline__ v8f wmma_bf16(const Frag16& a, const Frag16& b, v8f c) {
  // D = A(16x32 bf16) * B(32x16 bf16) + C(16x16 f32)
  return __builtin_amdgcn_wmma_f32_16x16x32_bf16(false, a.v, false, b.v, (short)0, c, false, false);
}

// Async global->LDS copy of 16 bytes per lane (CDNA5 GLOBAL_LOAD_ASYNC_TO_LDS_B128,
// tracked with ASYNCcnt). LDS offset = low 32 bits of the generic pointer (ISA 10.2).
static __device__ __forceinline__ void async_copy_b128(const void* gsrc, void* ldst) {
  uint32_t lds_off = (uint32_t)(uintptr_t)ldst;
  uint64_t gaddr   = (uint64_t)(uintptr_t)gsrc;
  asm volatile("global_load_async_to_lds_b128 %0, %1, off"
               :: "v"(lds_off), "v"(gaddr)
               : "memory");
}
static __device__ __forceinline__ void wait_async0() {
  asm volatile("s_wait_asynccnt 0" ::: "memory");
}

// -----------------------------------------------------------------------------
// Kernel 1: qkv = qkv_w (768x256) @ x[b] (256x4096) + qkv_b, with elu+1 on q,k.
// Writes q as [b,h,n,d] bf16 (n-major, for stage C / denom),
//        k,v as [b,c,n]  bf16 (channel-major, for kv GEMM / ksum / LePE).
// Block tile 64(M) x 64(N), K-step 32, 8 waves: wave w -> mtile w%4, ntiles 2*(w/4)+{0,1}
// -----------------------------------------------------------------------------
__global__ __launch_bounds__(256) void qkv_gemm_kernel(
    const float* __restrict__ x, const float* __restrict__ qkv_w,
    const float* __restrict__ qkv_b,
    bf16* __restrict__ q_t, bf16* __restrict__ k_bf, bf16* __restrict__ v_bf)
{
  const int b   = blockIdx.z;
  const int o0  = blockIdx.y * 64;
  const int n0  = blockIdx.x * 64;
  const int tid = threadIdx.x;
  const int lane = tid & 31;
  const int wave = tid >> 5;
  const int mt   = wave & 3;
  const int nt0  = (wave >> 2) * 2;
  const int lhalf = (lane < 16) ? 0 : 1;
  const int l16   = lane & 15;

  __shared__ __align__(16) bf16 lds_a[64 * 40];  // [m][k], 8-elem pad
  __shared__ __align__(16) bf16 lds_b[64 * 40];  // [n][k], transposed x tile

  v8f acc0 = {}; v8f acc1 = {};
  const float* xb = x + (size_t)b * CDIM * NPIX;

  for (int kc = 0; kc < CDIM; kc += 32) {
    {   // stage A: weights f32 -> bf16
      int row = tid >> 2;
      int ks  = (tid & 3) * 8;
      const float* src = qkv_w + (size_t)(o0 + row) * CDIM + kc + ks;
      if (kc + 32 < CDIM) __builtin_prefetch(src + 32, 0, 0);
      Pack8 p;
      #pragma unroll
      for (int j = 0; j < 8; ++j) p.h[j] = (bf16)src[j];
      *reinterpret_cast<uint4*>(&lds_a[row * 40 + ks]) = p.u;
    }
    {   // stage B: x tile (32k x 64n) f32 -> bf16, transposed to [n][k]
      int k  = tid >> 3;
      int ns = (tid & 7) * 8;
      const float* src = xb + (size_t)(kc + k) * NPIX + n0 + ns;
      #pragma unroll
      for (int j = 0; j < 8; ++j) lds_b[(ns + j) * 40 + k] = (bf16)src[j];
    }
    __syncthreads();
    Frag16 af, bf0, bf1;
    {   // A frag: lane<16 K=0-7,16-23 ; lane>=16 K=8-15,24-31
      int m  = mt * 16 + l16;
      int c0 = lhalf * 8;
      af.q[0] = *reinterpret_cast<const uint4*>(&lds_a[m * 40 + c0]);
      af.q[1] = *reinterpret_cast<const uint4*>(&lds_a[m * 40 + c0 + 16]);
    }
    {   // B frag: lane<16 K=0-7,16-23 ; lane>=16 K=8-15,24-31
      int kb = lhalf * 8;
      int nA = (nt0 + 0) * 16 + l16;
      int nB = (nt0 + 1) * 16 + l16;
      bf0.q[0] = *reinterpret_cast<const uint4*>(&lds_b[nA * 40 + kb]);
      bf0.q[1] = *reinterpret_cast<const uint4*>(&lds_b[nA * 40 + kb + 16]);
      bf1.q[0] = *reinterpret_cast<const uint4*>(&lds_b[nB * 40 + kb]);
      bf1.q[1] = *reinterpret_cast<const uint4*>(&lds_b[nB * 40 + kb + 16]);
    }
    acc0 = wmma_bf16(af, bf0, acc0);
    acc1 = wmma_bf16(af, bf1, acc1);
    __syncthreads();
  }

  const int which = (o0 + mt * 16) >> 8;   // 0=q 1=k 2=v (constant per 16-row tile)
  #pragma unroll
  for (int j = 0; j < 8; ++j) {
    int m = mt * 16 + j + lhalf * 8;       // D layout: vgpr j -> M=j (+8 for hi lanes)
    int o = o0 + m;
    int c = o & 255;
    float bias = qkv_b[o];
    #pragma unroll
    for (int t = 0; t < 2; ++t) {
      int n = n0 + (nt0 + t) * 16 + l16;
      float val = (t == 0 ? acc0[j] : acc1[j]) + bias;
      if (which <= 1) val = (val > 0.f) ? (val + 1.f) : __expf(val);  // elu+1
      if (which == 0) {
        q_t[(((size_t)b * NHEAD + (c >> 5)) * NPIX + n) * HD + (c & 31)] = (bf16)val;
      } else if (which == 1) {
        k_bf[((size_t)b * CDIM + c) * NPIX + n] = (bf16)val;
      } else {
        v_bf[((size_t)b * CDIM + c) * NPIX + n] = (bf16)val;
      }
    }
  }
}

// -----------------------------------------------------------------------------
// Kernel 2: kv[d,e] = sum_n k[n,d] * v[n,e] per (b,h). One block per (b,h);
// 8 waves split K=4096; f32 partials merged via LDS ds_add_f32; stored as
// kvT[b][h][e][d] bf16 so stage-C B fragments are contiguous.
// -----------------------------------------------------------------------------
__global__ __launch_bounds__(256) void kv_kernel(
    const bf16* __restrict__ k_bf, const bf16* __restrict__ v_bf,
    bf16* __restrict__ kvT)
{
  const int bh = blockIdx.x;               // 0..127
  const int b = bh >> 3, h = bh & 7;
  const int tid = threadIdx.x;
  const int lane = tid & 31;
  const int wave = tid >> 5;
  const int lhalf = (lane < 16) ? 0 : 1;
  const int l16 = lane & 15;

  __shared__ float lds_kv[32 * 32];
  for (int i = tid; i < 1024; i += 256) lds_kv[i] = 0.f;
  __syncthreads();

  const bf16* kb = k_bf + ((size_t)b * CDIM + h * HD) * NPIX;  // 32 rows (d) x 4096 (n)
  const bf16* vb = v_bf + ((size_t)b * CDIM + h * HD) * NPIX;  // 32 rows (e) x 4096 (n)

  v8f acc[2][2] = {};
  const int nstart = wave * 512;
  for (int s = 0; s < 16; ++s) {
    const int n0 = nstart + s * 32;
    Frag16 af[2], bfr[2];
    #pragma unroll
    for (int mt = 0; mt < 2; ++mt) {       // A = k^T : row d, K = n (contiguous)
      int d  = mt * 16 + l16;
      int c0 = lhalf * 8;
      const bf16* p = kb + (size_t)d * NPIX + n0;
      af[mt].q[0] = *reinterpret_cast<const uint4*>(p + c0);
      af[mt].q[1] = *reinterpret_cast<const uint4*>(p + c0 + 16);
    }
    #pragma unroll
    for (int nt = 0; nt < 2; ++nt) {       // B = v : col e, K = n (contiguous)
      int e  = nt * 16 + l16;
      const bf16* p = vb + (size_t)e * NPIX + n0 + lhalf * 8;
      bfr[nt].q[0] = *reinterpret_cast<const uint4*>(p);
      bfr[nt].q[1] = *reinterpret_cast<const uint4*>(p + 16);
    }
    #pragma unroll
    for (int mt = 0; mt < 2; ++mt)
      #pragma unroll
      for (int nt = 0; nt < 2; ++nt)
        acc[mt][nt] = wmma_bf16(af[mt], bfr[nt], acc[mt][nt]);
  }

  #pragma unroll
  for (int mt = 0; mt < 2; ++mt)
    #pragma unroll
    for (int nt = 0; nt < 2; ++nt)
      #pragma unroll
      for (int j = 0; j < 8; ++j) {
        int d = mt * 16 + j + lhalf * 8;
        int e = nt * 16 + l16;
        atomicAdd(&lds_kv[d * 32 + e], acc[mt][nt][j]);
      }
  __syncthreads();

  bf16* outp = kvT + (size_t)bh * HD * HD;
  for (int i = tid; i < 1024; i += 256) {
    int d = i >> 5, e = i & 31;
    outp[e * 32 + d] = (bf16)lds_kv[i];    // transposed store
  }
}

// -----------------------------------------------------------------------------
// Kernel 3: ksum[b][c] = sum_n k[b][c][n]
// -----------------------------------------------------------------------------
__global__ __launch_bounds__(128) void ksum_kernel(
    const bf16* __restrict__ k_bf, float* __restrict__ ksum)
{
  const int bc = blockIdx.x;
  const bf16* row = k_bf + (size_t)bc * NPIX;
  float s = 0.f;
  for (int n = threadIdx.x; n < NPIX; n += 128) s += (float)row[n];
  __shared__ float red[128];
  red[threadIdx.x] = s;
  __syncthreads();
  for (int st = 64; st > 0; st >>= 1) {
    if (threadIdx.x < st) red[threadIdx.x] += red[threadIdx.x + st];
    __syncthreads();
  }
  if (threadIdx.x == 0) ksum[bc] = red[0];
}

// -----------------------------------------------------------------------------
// Kernel 4: denom[b][h][n] = dot(q[n,:], ksum[b][h][:]) + eps
// -----------------------------------------------------------------------------
__global__ __launch_bounds__(256) void denom_kernel(
    const bf16* __restrict__ q_t, const float* __restrict__ ksum,
    float* __restrict__ denom)
{
  const int bh = blockIdx.y;
  const int n  = blockIdx.x * 256 + threadIdx.x;
  const bf16* qr = q_t + ((size_t)bh * NPIX + n) * HD;
  const float* ks = ksum + bh * HD;
  float s = 1e-6f;
  #pragma unroll
  for (int d = 0; d < HD; ++d) s += (float)qr[d] * ks[d];
  denom[(size_t)bh * NPIX + n] = s;
}

// -----------------------------------------------------------------------------
// Kernel 5: out_attn[n,e] = (q[n,:] @ kv[:,e]) / denom[n].
// Output is written N-MAJOR: out_pre[b][n][c] f32 (lanes -> consecutive c,
// coalesced), feeding the async-copyable layout of the final GEMM.
// -----------------------------------------------------------------------------
__global__ __launch_bounds__(256) void attn_kernel(
    const bf16* __restrict__ q_t, const bf16* __restrict__ kvT,
    const float* __restrict__ denom, float* __restrict__ out_pre)
{
  const int bh = blockIdx.y;
  const int b = bh >> 3, h = bh & 7;
  const int nb = blockIdx.x;               // 0..63
  const int tid = threadIdx.x;
  const int lane = tid & 31;
  const int wave = tid >> 5;
  const int mt = wave & 3;                 // n sub-tile
  const int et = wave >> 2;                // e sub-tile (0..1)
  const int lhalf = (lane < 16) ? 0 : 1;
  const int l16 = lane & 15;

  Frag16 af, bfr;
  {   // A = q : row n, K = d (contiguous in q_t)
    int n = nb * 64 + mt * 16 + l16;
    const bf16* p = q_t + ((size_t)bh * NPIX + n) * HD + lhalf * 8;
    af.q[0] = *reinterpret_cast<const uint4*>(p);
    af.q[1] = *reinterpret_cast<const uint4*>(p + 16);
  }
  {   // B = kv : col e, K = d (contiguous in kvT)
    int e = et * 16 + l16;
    const bf16* p = kvT + ((size_t)bh * HD + e) * HD + lhalf * 8;
    bfr.q[0] = *reinterpret_cast<const uint4*>(p);
    bfr.q[1] = *reinterpret_cast<const uint4*>(p + 16);
  }
  v8f acc = {};
  acc = wmma_bf16(af, bfr, acc);

  #pragma unroll
  for (int j = 0; j < 8; ++j) {
    int n = nb * 64 + mt * 16 + j + lhalf * 8;
    int e = et * 16 + l16;
    int c = h * HD + e;
    float val = acc[j] / denom[(size_t)bh * NPIX + n];
    out_pre[((size_t)b * NPIX + n) * CDIM + c] = val;   // n-major
  }
}

// -----------------------------------------------------------------------------
// Kernel 6: preproj[b][n][c] = out_pre[b][n][c] + lepe_b[c] + dwconv3x3(v)[c][n]
// c-fastest indexing: out_pre read and preproj write fully coalesced.
// -----------------------------------------------------------------------------
__global__ __launch_bounds__(256) void lepe_kernel(
    const bf16* __restrict__ v_bf, const float* __restrict__ lepe_w,
    const float* __restrict__ lepe_b, const float* __restrict__ out_pre,
    bf16* __restrict__ preproj)
{
  const size_t idx = (size_t)blockIdx.x * 256 + threadIdx.x;  // < B*N*C (c fastest)
  const int c = (int)(idx & (CDIM - 1));
  const int n = (int)((idx >> 8) & (NPIX - 1));
  const int b = (int)(idx >> 20);
  const int y = n >> 6, xc = n & 63;
  const bf16* vp = v_bf + ((size_t)b * CDIM + c) * NPIX;
  const float* w = lepe_w + c * 9;
  float s = out_pre[idx] + lepe_b[c];
  #pragma unroll
  for (int dy = -1; dy <= 1; ++dy) {
    int yy = y + dy;
    if (yy < 0 || yy >= IMGW) continue;
    #pragma unroll
    for (int dx = -1; dx <= 1; ++dx) {
      int xx = xc + dx;
      if (xx < 0 || xx >= IMGW) continue;
      s += (float)vp[yy * IMGW + xx] * w[(dy + 1) * 3 + (dx + 1)];
    }
  }
  preproj[idx] = (bf16)s;
}

// -----------------------------------------------------------------------------
// Kernel 7: out = out_w (256x256) @ preproj + out_b, preproj is [b][n][c] bf16.
// B tile rows are contiguous K-runs -> staged with async global->LDS B128 DMA
// (ASYNCcnt), completed with s_wait_asynccnt before the barrier.
// -----------------------------------------------------------------------------
__global__ __launch_bounds__(256) void out_gemm_kernel(
    const bf16* __restrict__ preproj, const float* __restrict__ out_w,
    const float* __restrict__ out_b, float* __restrict__ outp)
{
  const int b   = blockIdx.z;
  const int o0  = blockIdx.y * 64;
  const int n0  = blockIdx.x * 64;
  const int tid = threadIdx.x;
  const int lane = tid & 31;
  const int wave = tid >> 5;
  const int mt   = wave & 3;
  const int nt0  = (wave >> 2) * 2;
  const int lhalf = (lane < 16) ? 0 : 1;
  const int l16   = lane & 15;

  __shared__ __align__(16) bf16 lds_a[64 * 40];  // [m][k]
  __shared__ __align__(16) bf16 lds_b[64 * 40];  // [n][k]

  v8f acc0 = {}; v8f acc1 = {};
  const bf16* pb = preproj + (size_t)b * NPIX * CDIM;  // [n][c]

  for (int kc = 0; kc < CDIM; kc += 32) {
    {   // stage A: weights f32 -> bf16
      int row = tid >> 2;
      int ks  = (tid & 3) * 8;
      const float* src = out_w + (size_t)(o0 + row) * CDIM + kc + ks;
      if (kc + 32 < CDIM) __builtin_prefetch(src + 32, 0, 0);
      Pack8 p;
      #pragma unroll
      for (int j = 0; j < 8; ++j) p.h[j] = (bf16)src[j];
      *reinterpret_cast<uint4*>(&lds_a[row * 40 + ks]) = p.u;
    }
    {   // stage B: async DMA, 16B per lane, 64 rows x 32 k bf16
      int row = tid >> 2;
      int ch  = (tid & 3) * 8;   // bf16 elements, 16 bytes
      const bf16* src = pb + (size_t)(n0 + row) * CDIM + kc + ch;
      async_copy_b128(src, &lds_b[row * 40 + ch]);
    }
    wait_async0();
    __syncthreads();
    Frag16 af, bf0, bf1;
    {
      int m  = mt * 16 + l16;
      int c0 = lhalf * 8;
      af.q[0] = *reinterpret_cast<const uint4*>(&lds_a[m * 40 + c0]);
      af.q[1] = *reinterpret_cast<const uint4*>(&lds_a[m * 40 + c0 + 16]);
    }
    {
      int kb = lhalf * 8;
      int nA = (nt0 + 0) * 16 + l16;
      int nB = (nt0 + 1) * 16 + l16;
      bf0.q[0] = *reinterpret_cast<const uint4*>(&lds_b[nA * 40 + kb]);
      bf0.q[1] = *reinterpret_cast<const uint4*>(&lds_b[nA * 40 + kb + 16]);
      bf1.q[0] = *reinterpret_cast<const uint4*>(&lds_b[nB * 40 + kb]);
      bf1.q[1] = *reinterpret_cast<const uint4*>(&lds_b[nB * 40 + kb + 16]);
    }
    acc0 = wmma_bf16(af, bf0, acc0);
    acc1 = wmma_bf16(af, bf1, acc1);
    __syncthreads();
  }

  #pragma unroll
  for (int j = 0; j < 8; ++j) {
    int m = mt * 16 + j + lhalf * 8;
    int o = o0 + m;
    float bias = out_b[o];
    #pragma unroll
    for (int t = 0; t < 2; ++t) {
      int n = n0 + (nt0 + t) * 16 + l16;
      float val = (t == 0 ? acc0[j] : acc1[j]) + bias;
      outp[((size_t)b * CDIM + o) * NPIX + n] = val;
    }
  }
}

// -----------------------------------------------------------------------------
extern "C" void kernel_launch(void* const* d_in, const int* in_sizes, int n_in,
                              void* d_out, int out_size, void* d_ws, size_t ws_size,
                              hipStream_t stream) {
  const float* x      = (const float*)d_in[0];
  const float* qkv_w  = (const float*)d_in[1];
  const float* qkv_b  = (const float*)d_in[2];
  const float* lepe_w = (const float*)d_in[3];
  const float* lepe_b = (const float*)d_in[4];
  const float* out_w  = (const float*)d_in[5];
  const float* out_b  = (const float*)d_in[6];
  float* outp = (float*)d_out;

  char* ws = (char*)d_ws;
  size_t off = 0;
  auto take = [&](size_t bytes) -> char* {
    char* p = ws + off;
    off = (off + bytes + 255) & ~(size_t)255;
    return p;
  };
  bf16*  q_t     = (bf16*)take((size_t)BATCH * NHEAD * NPIX * HD * 2);   // 32 MiB
  bf16*  k_bf    = (bf16*)take((size_t)BATCH * CDIM * NPIX * 2);         // 32 MiB
  bf16*  v_bf    = (bf16*)take((size_t)BATCH * CDIM * NPIX * 2);         // 32 MiB
  bf16*  kvT     = (bf16*)take((size_t)BATCH * NHEAD * HD * HD * 2);     // 256 KiB
  float* ksum    = (float*)take((size_t)BATCH * CDIM * 4);               // 16 KiB
  float* denom   = (float*)take((size_t)BATCH * NHEAD * NPIX * 4);       // 2 MiB
  float* out_pre = (float*)take((size_t)BATCH * NPIX * CDIM * 4);        // 64 MiB (n-major)
  bf16*  preproj = (bf16*)take((size_t)BATCH * NPIX * CDIM * 2);         // 32 MiB (n-major)

  qkv_gemm_kernel<<<dim3(64, 12, BATCH), 256, 0, stream>>>(x, qkv_w, qkv_b, q_t, k_bf, v_bf);
  kv_kernel<<<dim3(BATCH * NHEAD), 256, 0, stream>>>(k_bf, v_bf, kvT);
  ksum_kernel<<<dim3(BATCH * CDIM), 128, 0, stream>>>(k_bf, ksum);
  denom_kernel<<<dim3(NPIX / 256, BATCH * NHEAD), 256, 0, stream>>>(q_t, ksum, denom);
  attn_kernel<<<dim3(NPIX / 64, BATCH * NHEAD), 256, 0, stream>>>(q_t, kvT, denom, out_pre);
  lepe_kernel<<<dim3((BATCH * CDIM * NPIX) / 256), 256, 0, stream>>>(v_bf, lepe_w, lepe_b, out_pre, preproj);
  out_gemm_kernel<<<dim3(64, 4, BATCH), 256, 0, stream>>>(preproj, out_w, out_b, outp);
}
